// MoeLayer_13932873908671
// MI455X (gfx1250) — matrix operand order repaired and checked by reading, not compiled, not run on hardware
//
#include <hip/hip_runtime.h>
#include <hip/hip_bf16.h>
#include <cstdint>

#if __has_include(<hip/amd_detail/amd_gfx1250_TDM.h>)
#define TDM_SIX_ARG 1
#else
#define TDM_SIX_ARG 0
#endif

#if __has_builtin(__builtin_amdgcn_tensor_load_to_lds)
#warning "PROBE: tensor_load_to_lds builtin PRESENT"
#if TDM_SIX_ARG
#warning "PROBE: using 6-arg TDM form (TDM header found)"
#else
#warning "PROBE: using 5-arg TDM form (no TDM header)"
#endif
#else
#warning "PROBE: tensor_load_to_lds builtin ABSENT -> cooperative fallback"
#endif

// ---------------- problem constants ----------------
constexpr int B_ = 4, S_ = 4096, H_ = 2048;
constexpr int E_ = 8, I_ = 1024;
constexpr int T_ = B_ * S_;          // 16384 tokens
constexpr int TILE_T = 48;           // tokens per FFN block (full-K X tile + H1 fit 320KB LDS)
constexpr int NWAVE = 16;            // 512 threads per block
constexpr int MAX_TILES = (T_ + TILE_T - 1) / TILE_T;   // 342

typedef __attribute__((ext_vector_type(16))) _Float16 v16h;
typedef __attribute__((ext_vector_type(8)))  _Float16 v8h;
typedef __attribute__((ext_vector_type(8)))  float    v8f;
typedef __attribute__((ext_vector_type(4)))  unsigned int u32x4;
typedef __attribute__((ext_vector_type(8)))  int      i32x8;
typedef __attribute__((ext_vector_type(4)))  int      i32x4;

union V16U { v16h v; v8h h[2]; };

__device__ __forceinline__ float gelu_exact(float x) {
    return 0.5f * x * (1.0f + erff(x * 0.70710678118654752f));
}

__device__ __forceinline__ v8f wmma_f16(v16h a, v16h b, v8f c) {
    return __builtin_amdgcn_wmma_f32_16x16x32_f16(false, a, false, b, (short)0, c, false, false);
}

// ---------------- workspace layout (bytes) ----------------
constexpr size_t WS_CNT   = 0;                                // E ints
constexpr size_t WS_TLIST = 4096;                             // E*T ints   (512 KB)
constexpr size_t WS_WLIST = WS_TLIST + (size_t)E_ * T_ * 4;   // E*T floats (512 KB)
constexpr size_t WS_XH    = 2u * 1024u * 1024u;               // T*H f16   (64 MB)
constexpr size_t WS_W1T   = WS_XH  + (size_t)T_ * H_ * 2;     // [E][I][H] f16 (32 MB)
constexpr size_t WS_W2T   = WS_W1T + (size_t)E_ * H_ * I_ * 2;// [E][H][I] f16 (32 MB)

// LDS layout in ffn_kernel (dynamic)
constexpr size_t LDS_X   = 0;                                  // [48][H] f16 = 192 KB
constexpr size_t LDS_H1  = (size_t)TILE_T * H_ * 2;            // [48][I] f16 = 96 KB
constexpr size_t LDS_TOK = LDS_H1 + (size_t)TILE_T * I_ * 2;
constexpr size_t LDS_W   = LDS_TOK + TILE_T * 4;
constexpr size_t LDS_BYTES = LDS_W + TILE_T * 4;               // ~288.5 KB

// ---------------- conversion kernels ----------------
__global__ void cvt_x_f16(const float* __restrict__ x, _Float16* __restrict__ xh, long n) {
    long i = (long)blockIdx.x * blockDim.x + threadIdx.x;
    long stride = (long)gridDim.x * blockDim.x;
    for (; i < n; i += stride) xh[i] = (_Float16)x[i];
}

// LDS-tiled transpose+convert: W1 [E][H][I] f32 -> W1T [E][I][H] f16
__global__ void cvt_w1t(const float* __restrict__ w1, _Float16* __restrict__ w1t) {
    __shared__ _Float16 tile[64][72];
    const int e = blockIdx.z, hb = blockIdx.y * 64, ib = blockIdx.x * 64;
    const int tid = threadIdx.x, r0 = tid >> 6, c = tid & 63;
    const float* src = w1 + ((size_t)e * H_ + hb) * I_ + ib;
#pragma unroll
    for (int rr = 0; rr < 64; rr += 4)
        tile[rr + r0][c] = (_Float16)src[(size_t)(rr + r0) * I_ + c];
    __syncthreads();
    _Float16* dst = w1t + ((size_t)e * I_ + ib) * H_ + hb;
#pragma unroll
    for (int rr = 0; rr < 64; rr += 4)
        dst[(size_t)(rr + r0) * H_ + c] = tile[c][rr + r0];
}

// LDS-tiled transpose+convert: W2 [E][I][H] f32 -> W2T [E][H][I] f16
__global__ void cvt_w2t(const float* __restrict__ w2, _Float16* __restrict__ w2t) {
    __shared__ _Float16 tile[64][72];
    const int e = blockIdx.z, ib = blockIdx.y * 64, hb = blockIdx.x * 64;
    const int tid = threadIdx.x, r0 = tid >> 6, c = tid & 63;
    const float* src = w2 + ((size_t)e * I_ + ib) * H_ + hb;
#pragma unroll
    for (int rr = 0; rr < 64; rr += 4)
        tile[rr + r0][c] = (_Float16)src[(size_t)(rr + r0) * H_ + c];
    __syncthreads();
    _Float16* dst = w2t + ((size_t)e * H_ + hb) * I_ + ib;
#pragma unroll
    for (int rr = 0; rr < 64; rr += 4)
        dst[(size_t)(rr + r0) * I_ + c] = tile[c][rr + r0];
}

// ---------------- gating kernel: one wave per token ----------------
__global__ void gate_kernel(const float* __restrict__ x, const float* __restrict__ wg,
                            int* __restrict__ cnt, int* __restrict__ tlist,
                            float* __restrict__ wlist) {
    int wave = threadIdx.x >> 5;
    int lane = threadIdx.x & 31;
    int t = blockIdx.x * 8 + wave;
    if (t >= T_) return;

    float acc[E_];
#pragma unroll
    for (int e = 0; e < E_; ++e) acc[e] = 0.0f;

    const float* xp = x + (size_t)t * H_;
    for (int h = lane; h < H_; h += 32) {
        float xv = xp[h];
        const float* wr = wg + (size_t)h * E_;
#pragma unroll
        for (int e = 0; e < E_; ++e) acc[e] += xv * wr[e];
    }
#pragma unroll
    for (int e = 0; e < E_; ++e) {
#pragma unroll
        for (int off = 16; off > 0; off >>= 1)
            acc[e] += __shfl_xor(acc[e], off, 32);
    }
    if (lane == 0) {
        int e0 = 0; float v0 = acc[0];
#pragma unroll
        for (int e = 1; e < E_; ++e) if (acc[e] > v0) { v0 = acc[e]; e0 = e; }
        int e1 = -1; float v1 = -3.4e38f;
#pragma unroll
        for (int e = 0; e < E_; ++e) if (e != e0 && acc[e] > v1) { v1 = acc[e]; e1 = e; }
        float ew = __expf(v1 - v0);
        float w0 = 1.0f / (1.0f + ew);
        float w1 = ew * w0;
        int s0 = atomicAdd(&cnt[e0], 1);
        tlist[(size_t)e0 * T_ + s0] = t;
        wlist[(size_t)e0 * T_ + s0] = w0;
        int s1 = atomicAdd(&cnt[e1], 1);
        tlist[(size_t)e1 * T_ + s1] = t;
        wlist[(size_t)e1 * T_ + s1] = w1;
    }
}

// TDM gather: load 16 token rows (H_ f16 elems each) of src into LDS at ldsoff.
__device__ __forceinline__ void tdm_gather16(const _Float16* src, unsigned int ldsoff,
                                             const int* toks) {
#if __has_builtin(__builtin_amdgcn_tensor_load_to_lds)
    unsigned long long ga = (unsigned long long)(size_t)src;
    u32x4 g0;
    g0[0] = 1u | (1u << 31);                     // count=1, 16-bit idx, gather_mode=1
    g0[1] = ldsoff;                              // lds_addr
    g0[2] = (unsigned int)(ga & 0xFFFFFFFFu);    // global_addr[31:0]
    g0[3] = (unsigned int)((ga >> 32) & 0x01FFFFFFu) | (2u << 30);  // addr[56:32] | type=2

    i32x8 g1;
    g1[0] = (1 << 16);                           // data_size = 1 (2 bytes)
    g1[1] = (H_ & 0xFFFF) << 16;                 // tensor_dim0[15:0] (=2048)
    g1[2] = (H_ >> 16) | ((T_ & 0xFFFF) << 16);  // tensor_dim0[31:16] | tensor_dim1[15:0]
    g1[3] = (T_ >> 16) | (H_ << 16);             // tensor_dim1[31:16] | tile_dim0 (=2048)
    g1[4] = 16;                                  // tile_dim1 = #indices, tile_dim2 = 0
    g1[5] = H_;                                  // tensor_dim0_stride[31:0]
    g1[6] = 0;                                   // stride0[47:32] | stride1[15:0]
    g1[7] = 0;                                   // stride1[47:16]

    i32x4 g2, g3;
#pragma unroll
    for (int j = 0; j < 4; ++j)
        g2[j] = (toks[2 * j] & 0xFFFF) | (toks[2 * j + 1] << 16);
#pragma unroll
    for (int j = 0; j < 4; ++j)
        g3[j] = (toks[8 + 2 * j] & 0xFFFF) | (toks[8 + 2 * j + 1] << 16);

#if TDM_SIX_ARG
    i32x8 gx = {0, 0, 0, 0, 0, 0, 0, 0};
    __builtin_amdgcn_tensor_load_to_lds(g0, g1, g2, g3, gx, 0);
#else
    __builtin_amdgcn_tensor_load_to_lds(g0, g1, g2, g3, 0);
#endif
#endif
}

// ---------------- fused expert FFN ----------------
// grid: (MAX_TILES, E). block: 512 threads (16 waves), one workgroup per WGP.
__global__ void __launch_bounds__(512)
ffn_kernel(const _Float16* __restrict__ xh,
           const _Float16* __restrict__ w1t,  // [E][I][H]
           const _Float16* __restrict__ w2t,  // [E][H][I]
           const float* __restrict__ b1,      // [E][I]
           const float* __restrict__ b2,      // [E][H]
           const int* __restrict__ cnt,
           const int* __restrict__ tlist,
           const float* __restrict__ wlist,
           float* __restrict__ out) {
    const int e = blockIdx.y;
    const int n = cnt[e];
    const int tile_base = blockIdx.x * TILE_T;
    if (tile_base >= n) return;

    extern __shared__ char smem[];
    _Float16* sX    = (_Float16*)(smem + LDS_X);    // [TILE_T][H_]
    _Float16* sH1   = (_Float16*)(smem + LDS_H1);   // [TILE_T][I_]
    int*      s_tok = (int*)(smem + LDS_TOK);
    float*    s_w   = (float*)(smem + LDS_W);

    const int tid   = threadIdx.x;
    const int wave  = tid >> 5;
    const int lane  = tid & 31;
    const int half  = lane >> 4;
    const int mlane = lane & 15;
    const int koffA = half * 8;
    const int koffB = half * 16;
    const int nvalid = (n - tile_base < TILE_T) ? (n - tile_base) : TILE_T;

    if (tid < TILE_T) {
        int idx = tile_base + tid;
        if (idx < n) {
            s_tok[tid] = tlist[(size_t)e * T_ + idx];
            s_w[tid]   = wlist[(size_t)e * T_ + idx];
        } else {
            s_tok[tid] = 0;
            s_w[tid]   = 0.0f;
        }
    }

#if __has_builtin(__builtin_amdgcn_tensor_load_to_lds)
    if (wave == 0) {
        unsigned int ldsbase = (unsigned int)(size_t)(void*)sX;
#pragma unroll
        for (int g = 0; g < 3; ++g) {
            int toks[16];
#pragma unroll
            for (int j = 0; j < 16; ++j) {
                int idx = tile_base + g * 16 + j;
                toks[j] = (idx < n) ? tlist[(size_t)e * T_ + idx] : 0;
            }
            tdm_gather16(xh, ldsbase + (unsigned int)(g * 16 * H_ * 2), toks);
        }
        __builtin_amdgcn_s_wait_tensorcnt(0);
    }
#else
    // fallback: cooperative gather, rows strided over 16 waves, v8h chunks
    {
        __syncthreads();
        for (int row = wave; row < TILE_T; row += NWAVE) {
            const _Float16* src = xh + (size_t)s_tok[row] * H_;
            _Float16* dst = sX + (size_t)row * H_;
            for (int kk = lane * 8; kk < H_; kk += 32 * 8)
                *(v8h*)(dst + kk) = *(const v8h*)(src + kk);
        }
    }
#endif
    __syncthreads();

    const v8f vzero = {0, 0, 0, 0, 0, 0, 0, 0};

    // ---------- Phase 1: H1 = gelu(X @ W1 + b1) in LDS ----------
    // 16 waves; wave owns ntiles [wave*4, wave*4+4) of I/16=64, in pairs; 3 M-blocks
    for (int p = 0; p < 2; ++p) {
        const int nt0 = wave * 4 + 2 * p;
        const int nt1 = nt0 + 1;
        const _Float16* pb0 = w1t + ((size_t)e * I_ + (size_t)(nt0 * 16 + mlane)) * H_;
        const _Float16* pb1 = pb0 + (size_t)16 * H_;

        v8f acc0[3], acc1[3];
#pragma unroll
        for (int mb = 0; mb < 3; ++mb) { acc0[mb] = vzero; acc1[mb] = vzero; }

        for (int k = 0; k < H_; k += 32) {
            v16h bv0 = *(const v16h*)(pb0 + k + koffB);
            v16h bv1 = *(const v16h*)(pb1 + k + koffB);
#pragma unroll
            for (int mb = 0; mb < 3; ++mb) {
                const _Float16* ph = sX + (size_t)(mb * 16 + mlane) * H_;
                V16U a;
                a.h[0] = *(const v8h*)(ph + k + koffB);
                a.h[1] = *(const v8h*)(ph + k + 8 + koffB);
                acc0[mb] = wmma_f16(a.v, bv0, acc0[mb]);
                acc1[mb] = wmma_f16(a.v, bv1, acc1[mb]);
            }
        }
        float bias0 = b1[(size_t)e * I_ + nt0 * 16 + mlane];
        float bias1 = b1[(size_t)e * I_ + nt1 * 16 + mlane];
#pragma unroll
        for (int mb = 0; mb < 3; ++mb) {
#pragma unroll
            for (int v = 0; v < 8; ++v) {
                int row = mb * 16 + half * 8 + v;
                float f0 = gelu_exact(acc0[mb][v] + bias0);
                float f1 = gelu_exact(acc1[mb][v] + bias1);
                sH1[(size_t)row * I_ + nt0 * 16 + mlane] = (_Float16)f0;
                sH1[(size_t)row * I_ + nt1 * 16 + mlane] = (_Float16)f1;
            }
        }
    }
    __syncthreads();

    // ---------- Phase 2: out += combine * (H1 @ W2 + b2) ----------
    // wave owns ntiles [wave*8, wave*8+8) of H/16=128
    for (int q = 0; q < 8; ++q) {
        const int nt = wave * 8 + q;
        const _Float16* pb = w2t + ((size_t)e * H_ + (size_t)(nt * 16 + mlane)) * I_;

        v8f acc[3];
#pragma unroll
        for (int mb = 0; mb < 3; ++mb) acc[mb] = vzero;

        for (int k = 0; k < I_; k += 32) {
            v16h bv = *(const v16h*)(pb + k + koffB);
#pragma unroll
            for (int mb = 0; mb < 3; ++mb) {
                const _Float16* ph = sH1 + (size_t)(mb * 16 + mlane) * I_;
                V16U a;
                a.h[0] = *(const v8h*)(ph + k + koffB);
                a.h[1] = *(const v8h*)(ph + k + 8 + koffB);
                acc[mb] = wmma_f16(a.v, bv, acc[mb]);
            }
        }
        const int col = nt * 16 + mlane;
        const float bias = b2[(size_t)e * H_ + col];
#pragma unroll
        for (int mb = 0; mb < 3; ++mb) {
#pragma unroll
            for (int v = 0; v < 8; ++v) {
                int row = mb * 16 + half * 8 + v;
                if (row < nvalid) {
                    float val = (acc[mb][v] + bias) * s_w[row];
                    unsafeAtomicAdd(out + (size_t)s_tok[row] * H_ + col, val);
                }
            }
        }
    }
}

// ---------------- launch ----------------
extern "C" void kernel_launch(void* const* d_in, const int* in_sizes, int n_in,
                              void* d_out, int out_size, void* d_ws, size_t ws_size,
                              hipStream_t stream) {
    const float* x   = (const float*)d_in[0];
    const float* wg  = (const float*)d_in[1];
    const float* W1  = (const float*)d_in[2];
    const float* b1  = (const float*)d_in[3];
    const float* W2  = (const float*)d_in[4];
    const float* b2  = (const float*)d_in[5];
    float* out = (float*)d_out;

    char* ws = (char*)d_ws;
    int*      cnt   = (int*)(ws + WS_CNT);
    int*      tlist = (int*)(ws + WS_TLIST);
    float*    wlist = (float*)(ws + WS_WLIST);
    _Float16* xh    = (_Float16*)(ws + WS_XH);
    _Float16* w1t   = (_Float16*)(ws + WS_W1T);
    _Float16* w2t   = (_Float16*)(ws + WS_W2T);

    hipMemsetAsync(cnt, 0, 4096, stream);
    hipMemsetAsync(d_out, 0, (size_t)out_size * sizeof(float), stream);

    cvt_x_f16<<<8192, 256, 0, stream>>>(x, xh, (long)T_ * H_);
    cvt_w1t<<<dim3(I_ / 64, H_ / 64, E_), 256, 0, stream>>>(W1, w1t);
    cvt_w2t<<<dim3(H_ / 64, I_ / 64, E_), 256, 0, stream>>>(W2, w2t);

    gate_kernel<<<T_ / 8, 256, 0, stream>>>(x, wg, cnt, tlist, wlist);

    dim3 grid(MAX_TILES, E_);
    ffn_kernel<<<grid, 512, LDS_BYTES, stream>>>(xh, w1t, w2t, b1, b2, cnt, tlist, wlist, out);
}
